// CTMHead_61048665145447
// MI455X (gfx1250) — matrix operand order repaired and checked by reading, not compiled, not run on hardware
//
#include <hip/hip_runtime.h>
#include <hip/hip_bf16.h>
#include <math.h>

// ---------------- shapes ----------------
#define Bq   128
#define Sq   512
#define Hq   512
#define Dq   256
#define Mq   25
#define DHq  32
#define Pq   512
#define Tq   10
#define L0q  1536
#define NHq  8
#define HDq  64

typedef _Float16 h16;
typedef __attribute__((ext_vector_type(16))) _Float16 v16h;
typedef __attribute__((ext_vector_type(8)))  _Float16 v8h;
typedef __attribute__((ext_vector_type(8)))  float    v8f;

typedef unsigned int u32x4 __attribute__((ext_vector_type(4)));
typedef int          i32x4 __attribute__((ext_vector_type(4)));
typedef int          i32x8 __attribute__((ext_vector_type(8)));

// =====================================================================
// Tensor Data Mover: DMA one 16-row x 32-col f32 tile (row stride K)
// from global memory into LDS at byte offset ldsOff. 2-D D# descriptor:
//   group0: count=1 | lds_addr | global_addr | type=2
//   group1: data_size=4B, tensor_dim0=K, tensor_dim1=16,
//           tile_dim0=32, tile_dim1=16, tensor_dim0_stride=K
// Tracked by TENSORcnt.
// =====================================================================
__device__ __forceinline__ void tdm_load_tile(const float* gptr, unsigned ldsOff, int K)
{
    unsigned long long ga = (unsigned long long)(uintptr_t)gptr;
    u32x4 g0;
    g0[0] = 1u;                                             // count=1, user mode
    g0[1] = ldsOff;                                         // LDS byte address
    g0[2] = (unsigned)(ga & 0xffffffffu);                   // global addr lo
    g0[3] = (unsigned)((ga >> 32) & 0x01ffffffu) | 0x80000000u; // addr hi | type=2
    i32x8 g1;
    g1[0] = (int)(2u << 16);                                // data_size = 4 bytes
    g1[1] = (int)(((unsigned)K & 0xffffu) << 16);           // tensor_dim0[15:0]
    g1[2] = (int)((16u << 16) | ((unsigned)K >> 16));       // tensor_dim1=16 | dim0 hi
    g1[3] = (int)(32u << 16);                               // tile_dim0 = 32
    g1[4] = 16;                                             // tile_dim1 = 16
    g1[5] = K;                                              // tensor_dim0_stride lo
    g1[6] = 0;
    g1[7] = 0;
    i32x4 gz = {0, 0, 0, 0};
#if defined(__clang_major__) && (__clang_major__ >= 23)
    i32x8 gz8 = {0, 0, 0, 0, 0, 0, 0, 0};
    __builtin_amdgcn_tensor_load_to_lds(g0, g1, gz, gz, gz8, 0);
#else
    __builtin_amdgcn_tensor_load_to_lds(g0, g1, gz, gz, 0);
#endif
}

// =====================================================================
// Generic WMMA GEMM:  C[M,N] = A[M,K](f32) * W[N,K]^T (f16) + bias
// block = 128 thr (4 waves); block tile = 16 rows x 512 cols
// grid  = (M/16, ceil(N/512))
// A tile staged to LDS by the TDM (double buffered), overlapping the
// 8 back-to-back v_wmma of the previous chunk. Inner loop is branchless.
// =====================================================================
__launch_bounds__(128)
__global__ void k_wmma_gemm(const float* __restrict__ A, int K,
                            const h16* __restrict__ Wt,
                            const float* __restrict__ bias,
                            float* __restrict__ Cf, h16* __restrict__ Ch,
                            int N)
{
    __shared__ float ldsA[2][16 * 32];

    const int rowBase  = blockIdx.x * 16;
    const int colBlock = blockIdx.y * 512;
    const int tid  = threadIdx.x;
    const int lane = tid & 31;
    const int wid  = tid >> 5;
    const int r    = lane & 15;
    const int hi   = lane >> 4;

    const unsigned ldsOff0 = (unsigned)(uintptr_t)&ldsA[0][0];
    const unsigned ldsOff1 = (unsigned)(uintptr_t)&ldsA[1][0];
    const float* Abase = A + (size_t)rowBase * (size_t)K;
    const int nch = K >> 5;                      // K chunks of 32

    v8f acc[8] = {};

    if (wid == 0) tdm_load_tile(Abase, ldsOff0, K);   // prologue: chunk 0

    for (int i = 0; i < nch; i++) {
        if (wid == 0) {
            if (i + 1 < nch) {
                // issue next chunk into the other buffer, then wait for
                // the current chunk only (in-order TDM completion)
                tdm_load_tile(Abase + (size_t)(i + 1) * 32,
                              ((i & 1) ? ldsOff0 : ldsOff1), K);
                __builtin_amdgcn_s_wait_tensorcnt(1);
            } else {
                __builtin_amdgcn_s_wait_tensorcnt(0);
            }
        }
        __syncthreads();                           // chunk i visible to all waves

        const float* lb = ldsA[i & 1];
        const float* p0 = lb + r * 32 + hi * 16;
        const float* p1 = lb + r * 32 + hi * 16 + 8;
        v16h af;
        #pragma unroll
        for (int j = 0; j < 8; j++) {
            af[j]     = (h16)p0[j];
            af[j + 8] = (h16)p1[j];
        }

        const int kk = i << 5;
        v16h bf[8];
        #pragma unroll
        for (int ct = 0; ct < 8; ct++) {
            int cb = colBlock + wid * 128 + ct * 16;
            cb = (cb <= N - 16) ? cb : (N - 16);   // branchless clamp
            bf[ct] = *(const v16h*)(Wt + (size_t)(cb + r) * (size_t)K + kk + hi * 16);
        }
        #pragma unroll
        for (int ct = 0; ct < 8; ct++)
            acc[ct] = __builtin_amdgcn_wmma_f32_16x16x32_f16(
                false, af, false, bf[ct], (short)0, acc[ct], false, false);

        __syncthreads();                           // buffer may be reused next iter
    }

    #pragma unroll
    for (int ct = 0; ct < 8; ct++) {
        int cb  = colBlock + wid * 128 + ct * 16;
        int col = cb + r;
        if (cb >= N) continue;                     // epilogue only
        float bv = bias ? bias[col] : 0.0f;
        #pragma unroll
        for (int g = 0; g < 8; g++) {
            int row = rowBase + g + hi * 8;
            float val = acc[ct][g] + bv;
            if (Cf) Cf[(size_t)row * N + col] = val;
            if (Ch) Ch[(size_t)row * N + col] = (h16)val;
        }
    }
}

// =====================================================================
// LayerNorm (optionally followed by exact GeLU). One block per row.
// =====================================================================
__launch_bounds__(256)
__global__ void k_ln(const float* __restrict__ X, float* __restrict__ Y, int Wd,
                     const float* __restrict__ g, const float* __restrict__ bt,
                     int dogelu)
{
    __shared__ float s1[256], s2[256];
    const int row = blockIdx.x, tid = threadIdx.x;
    const float* xr = X + (size_t)row * Wd;
    float sum = 0.f, sq = 0.f;
    for (int i = tid; i < Wd; i += 256) { float v = xr[i]; sum += v; sq += v * v; }
    s1[tid] = sum; s2[tid] = sq; __syncthreads();
    for (int off = 128; off > 0; off >>= 1) {
        if (tid < off) { s1[tid] += s1[tid + off]; s2[tid] += s2[tid + off]; }
        __syncthreads();
    }
    float m   = s1[0] / Wd;
    float var = s2[0] / Wd - m * m;
    float rs  = rsqrtf(var + 1e-5f);
    float* yr = Y + (size_t)row * Wd;
    for (int i = tid; i < Wd; i += 256) {
        float v = (xr[i] - m) * rs * g[i] + bt[i];
        if (dogelu) v = 0.5f * v * (1.0f + erff(v * 0.70710678118f));
        yr[i] = v;
    }
}

// =====================================================================
// Attention for one (b,h): scores over S=512, softmax, weighted V sum.
// k/v are f16 (L2 resident). 128 threads/block, grid = B*NH.
// =====================================================================
__launch_bounds__(128)
__global__ void k_attn(const float* __restrict__ qh, const h16* __restrict__ K16,
                       const h16* __restrict__ V16, float* __restrict__ outp)
{
    __shared__ float qv[HDq];
    __shared__ float sc[Sq];
    __shared__ float red[128];
    const int bh = blockIdx.x, b = bh >> 3, h = bh & 7;
    const int tid = threadIdx.x;
    if (tid < HDq) qv[tid] = qh[(size_t)b * Hq + h * HDq + tid];
    __syncthreads();

    float lmax = -1e30f;
    #pragma unroll
    for (int j = 0; j < 4; j++) {
        int s = tid + j * 128;
        const h16* kp = K16 + ((size_t)b * Sq + s) * Hq + h * HDq;
        float a = 0.f;
        #pragma unroll 8
        for (int d = 0; d < HDq; d++) a += qv[d] * (float)kp[d];
        a *= 0.125f;            // 1/sqrt(64)
        sc[s] = a;
        lmax = fmaxf(lmax, a);
    }
    red[tid] = lmax; __syncthreads();
    for (int off = 64; off > 0; off >>= 1) {
        if (tid < off) red[tid] = fmaxf(red[tid], red[tid + off]);
        __syncthreads();
    }
    float mx = red[0]; __syncthreads();

    float lsum = 0.f;
    #pragma unroll
    for (int j = 0; j < 4; j++) {
        int s = tid + j * 128;
        float e = expf(sc[s] - mx);
        sc[s] = e; lsum += e;
    }
    red[tid] = lsum; __syncthreads();
    for (int off = 64; off > 0; off >>= 1) {
        if (tid < off) red[tid] += red[tid + off];
        __syncthreads();
    }
    float inv = 1.0f / red[0]; __syncthreads();

    const int d = tid & 63, part = tid >> 6;
    float o = 0.f;
    for (int s = part * 256; s < part * 256 + 256; s++)
        o += sc[s] * (float)V16[((size_t)b * Sq + s) * Hq + h * HDq + d];
    red[tid] = o; __syncthreads();
    if (part == 0)
        outp[(size_t)b * Hq + h * HDq + d] = (red[tid] + red[tid + 64]) * inv;
}

// ---------------- small helper kernels ----------------
__global__ void k_zero(float* p, int n)
{ int i = blockIdx.x * 256 + threadIdx.x; if (i < n) p[i] = 0.f; }

__global__ void k_cvt16(const float* __restrict__ w, h16* __restrict__ o, int n)
{ int i = blockIdx.x * 256 + threadIdx.x; if (i < n) o[i] = (h16)w[i]; }

__global__ void k_absmean(const float* __restrict__ w, float* acc, int n)
{
    __shared__ float s[256];
    float v = 0.f;
    for (int i = blockIdx.x * 256 + threadIdx.x; i < n; i += gridDim.x * 256) {
        float x = fminf(fmaxf(w[i], -2.f), 2.f); v += fabsf(x);
    }
    s[threadIdx.x] = v; __syncthreads();
    for (int off = 128; off > 0; off >>= 1) {
        if (threadIdx.x < off) s[threadIdx.x] += s[threadIdx.x + off];
        __syncthreads();
    }
    if (threadIdx.x == 0) atomicAdd(acc, s[0]);
}

__global__ void k_bitquant(const float* __restrict__ w, const float* __restrict__ acc,
                           h16* __restrict__ o, int n, float invN)
{
    int i = blockIdx.x * 256 + threadIdx.x; if (i >= n) return;
    float gamma = fmaxf(acc[0] * invN, 1e-4f);
    float x = fminf(fmaxf(w[i], -2.f), 2.f);
    float q = fminf(fmaxf(rintf(x / gamma), -1.f), 1.f) * gamma;
    o[i] = (h16)q;
}

__global__ void k_gates(const float* __restrict__ r, float* __restrict__ g, int n)
{
    int i = blockIdx.x * 256 + threadIdx.x; if (i >= n) return;
    float sp = log1pf(expf(r[i]));            // softplus
    g[i] = expf(-sp);
}

__global__ void k_bcast_z(const float* __restrict__ zi, float* __restrict__ z)
{ int i = blockIdx.x * 256 + threadIdx.x; if (i < Bq * Dq) z[i] = zi[i % Dq]; }

__global__ void k_sync(const float* __restrict__ z, const int* __restrict__ il,
                       const int* __restrict__ ir, const float* __restrict__ g,
                       float* __restrict__ Aa, float* __restrict__ Bb,
                       float* __restrict__ syncf)
{
    int i = blockIdx.x * 256 + threadIdx.x; if (i >= Bq * Pq) return;
    int b = i >> 9, p = i & 511;
    float gg   = g[p];
    float prod = z[b * Dq + il[p]] * z[b * Dq + ir[p]];
    float a = gg * Aa[i] + prod;  Aa[i] = a;
    float c = gg * Bb[i] + 1.0f;  Bb[i] = c;
    syncf[i] = a * rsqrtf(fmaxf(c, 1e-8f));
}

__global__ void k_concat(const float* __restrict__ z, const float* __restrict__ o,
                         float* __restrict__ cat)
{
    int i = blockIdx.x * 256 + threadIdx.x; if (i >= Bq * (Dq + Hq)) return;
    int b = i / (Dq + Hq), j = i % (Dq + Hq);
    cat[i] = (j < Dq) ? z[b * Dq + j] : o[b * Hq + (j - Dq)];
}

__global__ void k_hist(float* __restrict__ ah, const float* __restrict__ a,
                       const float* __restrict__ init, int first)
{
    int i = blockIdx.x * 256 + threadIdx.x; if (i >= Bq * Dq) return;
    int d = i & 255;
    float rbuf[Mq];
    const float* src = first ? (init + (size_t)d * Mq) : (ah + (size_t)i * Mq);
    #pragma unroll
    for (int m = 0; m < Mq; m++) rbuf[m] = src[m];
    float* dst = ah + (size_t)i * Mq;
    #pragma unroll
    for (int m = 0; m < Mq - 1; m++) dst[m] = rbuf[m + 1];
    dst[Mq - 1] = a[i];
}

__global__ void k_nlm(const float* __restrict__ ah, const float* __restrict__ w1,
                      const float* __restrict__ b1, const float* __restrict__ w2,
                      const float* __restrict__ b2, float* __restrict__ z)
{
    int i = blockIdx.x * 256 + threadIdx.x; if (i >= Bq * Dq) return;
    int d = i & 255;
    const float* ar = ah + (size_t)i * Mq;
    float acc = 0.f;
    for (int h = 0; h < DHq; h++) {
        float s = b1[h * Dq + d];
        #pragma unroll
        for (int m = 0; m < Mq; m++) s += ar[m] * w1[((size_t)m * DHq + h) * Dq + d];
        s = 0.5f * s * (1.0f + erff(s * 0.70710678118f));
        acc += s * w2[h * Dq + d];
    }
    z[i] = acc + b2[d];
}

__global__ void k_rowdot(const float* __restrict__ X, const float* __restrict__ w,
                         const float* __restrict__ b, float* __restrict__ o, int dosig)
{
    int r = blockIdx.x * blockDim.x + threadIdx.x; if (r >= Bq) return;
    float acc = b[0];
    const float* x = X + (size_t)r * Hq;
    for (int i = 0; i < Hq; i++) acc += x[i] * w[i];
    if (dosig) acc = 1.0f / (1.0f + expf(-acc));
    o[r] = acc;
}

// =====================================================================
// host-side orchestration
// =====================================================================
extern "C" void kernel_launch(void* const* d_in, const int* in_sizes, int n_in,
                              void* d_out, int out_size, void* d_ws, size_t ws_size,
                              hipStream_t stream)
{
    (void)in_sizes; (void)n_in; (void)out_size; (void)ws_size;

    const float* x        = (const float*)d_in[0];
    const int*   ial      = (const int*)d_in[1];
    const int*   iar      = (const int*)d_in[2];
    const int*   iol      = (const int*)d_in[3];
    const int*   ior      = (const int*)d_in[4];
    const float* z_init   = (const float*)d_in[5];
    const float* ah_init  = (const float*)d_in[6];
    const float* syn_w1   = (const float*)d_in[7];
    const float* syn_b1   = (const float*)d_in[8];
    const float* syn_ln1g = (const float*)d_in[9];
    const float* syn_ln1b = (const float*)d_in[10];
    const float* syn_w2   = (const float*)d_in[11];
    const float* syn_b2   = (const float*)d_in[12];
    const float* syn_ln2g = (const float*)d_in[13];
    const float* syn_ln2b = (const float*)d_in[14];
    const float* syn_w3   = (const float*)d_in[15];
    const float* syn_b3   = (const float*)d_in[16];
    const float* nlm_w1   = (const float*)d_in[17];
    const float* nlm_b1   = (const float*)d_in[18];
    const float* nlm_w2   = (const float*)d_in[19];
    const float* nlm_b2   = (const float*)d_in[20];
    const float* r_a      = (const float*)d_in[21];
    const float* r_o      = (const float*)d_in[22];
    const float* qp_w     = (const float*)d_in[23];
    const float* qp_b     = (const float*)d_in[24];
    const float* qp_lng   = (const float*)d_in[25];
    const float* qp_lnb   = (const float*)d_in[26];
    const float* kvp_w    = (const float*)d_in[27];
    const float* kvp_b    = (const float*)d_in[28];
    const float* kvp_lng  = (const float*)d_in[29];
    const float* kvp_lnb  = (const float*)d_in[30];
    const float* attn_win = (const float*)d_in[31];
    const float* attn_bin = (const float*)d_in[32];
    const float* attn_wo  = (const float*)d_in[33];
    const float* attn_bo  = (const float*)d_in[34];
    const float* cert_w   = (const float*)d_in[35];
    const float* cert_b   = (const float*)d_in[36];
    const float* val_w    = (const float*)d_in[37];
    const float* val_b    = (const float*)d_in[38];
    const float* op_w1    = (const float*)d_in[39];
    const float* op_b1    = (const float*)d_in[40];
    const float* op_ln1g  = (const float*)d_in[41];
    const float* op_ln1b  = (const float*)d_in[42];
    const float* op_w2    = (const float*)d_in[43];
    const float* op_b2    = (const float*)d_in[44];
    const float* op_ln2g  = (const float*)d_in[45];
    const float* op_ln2b  = (const float*)d_in[46];

    float* out   = (float*)d_out;
    float* featO = out;                        // B*L0
    float* syncO = out + (size_t)Bq * L0q;     // B*P
    float* certO = syncO + (size_t)Bq * Pq;    // B
    float* valO  = certO + Bq;                 // B

    // ---- bump allocator on workspace ----
    char* wptr = (char*)d_ws;
    auto alloc = [&](size_t bytes) -> void* {
        void* p = (void*)wptr;
        wptr += (bytes + 255) & ~(size_t)255;
        return p;
    };

    // zero-group (must be contiguous): aa, ba, ao, bo, gamma accs
    float* aa   = (float*)alloc((size_t)Bq * Pq * 4);
    float* ba   = (float*)alloc((size_t)Bq * Pq * 4);
    float* ao   = (float*)alloc((size_t)Bq * Pq * 4);
    float* bo   = (float*)alloc((size_t)Bq * Pq * 4);
    float* gacc = (float*)alloc(256);          // [0]=syn1 [1]=syn2 [2]=syn3
    const int zeroN = 4 * Bq * Pq + 64;

    float* ga    = (float*)alloc(Pq * 4);
    float* go    = (float*)alloc(Pq * 4);
    float* zbuf  = (float*)alloc((size_t)Bq * Dq * 4);
    float* ahist = (float*)alloc((size_t)Bq * Dq * Mq * 4);
    float* syncA = (float*)alloc((size_t)Bq * Pq * 4);
    float* qpre  = (float*)alloc((size_t)Bq * Hq * 4);
    float* qhb   = (float*)alloc((size_t)Bq * Hq * 4);
    float* atto  = (float*)alloc((size_t)Bq * Hq * 4);
    float* obuf  = (float*)alloc((size_t)Bq * Hq * 4);
    float* catb  = (float*)alloc((size_t)Bq * (Dq + Hq) * 4);
    float* h1b   = (float*)alloc((size_t)Bq * Hq * 4);
    float* h2b   = (float*)alloc((size_t)Bq * Hq * 4);
    float* abuf  = (float*)alloc((size_t)Bq * Dq * 4);
    float* hfb   = (float*)alloc((size_t)Bq * Hq * 4);
    float* fpre  = (float*)alloc((size_t)Bq * L0q * 4);

    h16* kvp_h  = (h16*)alloc((size_t)Hq * L0q * 2);
    h16* win_h  = (h16*)alloc((size_t)3 * Hq * Hq * 2);
    h16* wo_h   = (h16*)alloc((size_t)Hq * Hq * 2);
    h16* qp_h   = (h16*)alloc((size_t)Hq * Pq * 2);
    h16* op1_h  = (h16*)alloc((size_t)Hq * Pq * 2);
    h16* op2_h  = (h16*)alloc((size_t)L0q * Hq * 2);
    h16* syn1_h = (h16*)alloc((size_t)Hq * (Dq + Hq) * 2);
    h16* syn2_h = (h16*)alloc((size_t)Hq * Hq * 2);
    h16* syn3_h = (h16*)alloc((size_t)Dq * Hq * 2);

    float* kvbuf = (float*)alloc((size_t)Bq * Sq * Hq * 4);  // 134 MB
    h16*   k16   = (h16*)alloc((size_t)Bq * Sq * Hq * 2);    // 67 MB
    h16*   v16   = (h16*)alloc((size_t)Bq * Sq * Hq * 2);    // 67 MB

    auto blocks = [](int n) { return (n + 255) / 256; };

    auto gemm = [&](const float* A, int K, const h16* Wt, const float* bias,
                    float* Cf, h16* Ch, int Mrows, int N) {
        dim3 grid(Mrows / 16, (N + 511) / 512);
        k_wmma_gemm<<<grid, dim3(128), 0, stream>>>(A, K, Wt, bias, Cf, Ch, N);
    };
    auto ln = [&](const float* X, float* Y, int rows, int Wd,
                  const float* g, const float* b, int gelu) {
        k_ln<<<rows, 256, 0, stream>>>(X, Y, Wd, g, b, gelu);
    };
    auto cvt = [&](const float* w, h16* o, int n) {
        k_cvt16<<<blocks(n), 256, 0, stream>>>(w, o, n);
    };
    auto bitq = [&](const float* w, float* acc, h16* o, int n) {
        int g = blocks(n); if (g > 1024) g = 1024;
        k_absmean<<<g, 256, 0, stream>>>(w, acc, n);
        k_bitquant<<<blocks(n), 256, 0, stream>>>(w, acc, o, n, 1.0f / (float)n);
    };

    // ---- setup ----
    k_zero<<<blocks(zeroN), 256, 0, stream>>>(aa, zeroN);
    k_gates<<<blocks(Pq), 256, 0, stream>>>(r_a, ga, Pq);
    k_gates<<<blocks(Pq), 256, 0, stream>>>(r_o, go, Pq);
    k_bcast_z<<<blocks(Bq * Dq), 256, 0, stream>>>(z_init, zbuf);

    cvt(kvp_w, kvp_h, Hq * L0q);
    cvt(attn_win, win_h, 3 * Hq * Hq);
    cvt(attn_wo, wo_h, Hq * Hq);
    cvt(qp_w, qp_h, Hq * Pq);
    cvt(op_w1, op1_h, Hq * Pq);
    cvt(op_w2, op2_h, L0q * Hq);
    bitq(syn_w1, gacc + 0, syn1_h, Hq * (Dq + Hq));
    bitq(syn_w2, gacc + 1, syn2_h, Hq * Hq);
    bitq(syn_w3, gacc + 2, syn3_h, Dq * Hq);

    // ---- big projections (WMMA + TDM) ----
    gemm(x, L0q, kvp_h, kvp_b, kvbuf, nullptr, Bq * Sq, Hq);          // kv pre-LN
    ln(kvbuf, kvbuf, Bq * Sq, Hq, kvp_lng, kvp_lnb, 0);               // kv = LN(...)
    gemm(kvbuf, Hq, win_h + (size_t)Hq * Hq, attn_bin + Hq,
         nullptr, k16, Bq * Sq, Hq);                                  // k (f16)
    gemm(kvbuf, Hq, win_h + (size_t)2 * Hq * Hq, attn_bin + 2 * Hq,
         nullptr, v16, Bq * Sq, Hq);                                  // v (f16)

    // ---- scan over T ticks ----
    for (int t = 0; t < Tq; t++) {
        k_sync<<<blocks(Bq * Pq), 256, 0, stream>>>(zbuf, ial, iar, ga, aa, ba, syncA);
        gemm(syncA, Pq, qp_h, qp_b, qpre, nullptr, Bq, Hq);
        ln(qpre, qpre, Bq, Hq, qp_lng, qp_lnb, 0);
        gemm(qpre, Hq, win_h, attn_bin, qhb, nullptr, Bq, Hq);        // wq
        k_attn<<<Bq * NHq, 128, 0, stream>>>(qhb, k16, v16, atto);
        gemm(atto, Hq, wo_h, attn_bo, obuf, nullptr, Bq, Hq);
        if (t == Tq - 1) {
            k_rowdot<<<1, 128, 0, stream>>>(obuf, cert_w, cert_b, certO, 1);
            k_rowdot<<<1, 128, 0, stream>>>(obuf, val_w, val_b, valO, 0);
        }
        k_concat<<<blocks(Bq * (Dq + Hq)), 256, 0, stream>>>(zbuf, obuf, catb);
        gemm(catb, Dq + Hq, syn1_h, syn_b1, h1b, nullptr, Bq, Hq);
        ln(h1b, h1b, Bq, Hq, syn_ln1g, syn_ln1b, 1);
        gemm(h1b, Hq, syn2_h, syn_b2, h2b, nullptr, Bq, Hq);
        ln(h2b, h2b, Bq, Hq, syn_ln2g, syn_ln2b, 1);
        gemm(h2b, Hq, syn3_h, syn_b3, abuf, nullptr, Bq, Dq);
        k_hist<<<blocks(Bq * Dq), 256, 0, stream>>>(ahist, abuf, ah_init, t == 0);
        k_nlm<<<blocks(Bq * Dq), 256, 0, stream>>>(ahist, nlm_w1, nlm_b1,
                                                   nlm_w2, nlm_b2, zbuf);
        k_sync<<<blocks(Bq * Pq), 256, 0, stream>>>(zbuf, iol, ior, go, ao, bo, syncO);
    }

    // ---- output head ----
    gemm(syncO, Pq, op1_h, op_b1, hfb, nullptr, Bq, Hq);
    ln(hfb, hfb, Bq, Hq, op_ln1g, op_ln1b, 1);
    gemm(hfb, Hq, op2_h, op_b2, fpre, nullptr, Bq, L0q);
    ln(fpre, featO, Bq, L0q, op_ln2g, op_ln2b, 0);
}
